// TrajectoryPrediction_16355235463636
// MI455X (gfx1250) — matrix operand order repaired and checked by reading, not compiled, not run on hardware
//
#include <hip/hip_runtime.h>
#include <hip/hip_bf16.h>
#include <math.h>

// ---------------- problem constants ----------------
#define OBS   8
#define PRED  12
#define EIN   64
#define EHID  256
#define NOISE 64
#define GEMB  64
#define DHID  (EHID + NOISE)   // 320
#define GRP   8
#define NPER  192
#define BATCH (GRP * NPER)     // 1536
#define MAX_KT (EIN + DHID)    // 384 (largest combined K)

typedef __attribute__((ext_vector_type(16))) _Float16 v16h;
typedef __attribute__((ext_vector_type(8)))  float    v8f;

// ---------------- small device helpers ----------------
__device__ __forceinline__ float sigmoidf_(float x) { return 1.f / (1.f + __expf(-x)); }
__device__ __forceinline__ float leakyf_(float x)   { return x > 0.f ? x : 0.2f * x; }

__device__ __forceinline__ float blockReduceSum(float v, float* s) {
    int t = threadIdx.x;
    s[t] = v; __syncthreads();
    for (int off = blockDim.x >> 1; off > 0; off >>= 1) {
        if (t < off) s[t] += s[t + off];
        __syncthreads();
    }
    float r = s[0]; __syncthreads();
    return r;
}
__device__ __forceinline__ float blockReduceMax(float v, float* s) {
    int t = threadIdx.x;
    s[t] = v; __syncthreads();
    for (int off = blockDim.x >> 1; off > 0; off >>= 1) {
        if (t < off) s[t] = fmaxf(s[t], s[t + off]);
        __syncthreads();
    }
    float r = s[0]; __syncthreads();
    return r;
}

// B-fragment gather from the LDS weight panel (layout per ISA 7.12.2, wave32):
//   B 32x16 f16 : lane l (n=l&15, g=l>>4): v[i<4] -> K=2i+8g ; v[i>=4] -> K=16+2(i-4)+8g
__device__ __forceinline__ v16h load_bfrag(const _Float16* __restrict__ wb, int g) {
    v16h b;
#pragma unroll
    for (int i = 0; i < 8; ++i) {
        int kw = 2 * (i & 3) + 8 * g + 16 * (i >> 2);
        b[2 * i]     = wb[kw];
        b[2 * i + 1] = wb[kw + 1];
    }
    return b;
}

// ---------------- blocked WMMA GEMM ----------------
// Computes C[M,N] = X @ Wih^T + H @ Whh^T (+bias), all fp32 in memory, f16 WMMA math.
// Grid: (N/64, M/128). Block: 256 threads = 8 waves (wave32).
// Block stages the 64-column weight panel (both segments, f16) into LDS once
// (<= 64*384*2B = 48KB << 320KB/WGP). Each wave computes a 16x64 strip.
// Per k-step: one A fragment (global), then ALL FOUR B fragments pinned live via
// an asm register barrier (one ds clause + single s_wait_dscnt), then a 4-deep
// back-to-back WMMA burst (XDL issue pattern per flow/sync 7.5).
__global__ void k_gemm_block(const float* __restrict__ X, int Kx,
                             const float* __restrict__ Wih,
                             const float* __restrict__ H, int Kh,
                             const float* __restrict__ Whh,
                             const float* __restrict__ bias,
                             float* __restrict__ C, int M, int N) {
    __shared__ _Float16 Ws[64 * MAX_KT];     // [n][k], k contiguous
    const int KT = Kx + Kh;
    const int n0 = blockIdx.x * 64;
    const int mBase = blockIdx.y * 128;

    // ---- stage 64 x KT weight panel into LDS as f16 ----
    {
        const int kq = KT >> 2;              // float4 chunks per row
        for (int c = threadIdx.x; c < 64 * kq; c += 256) {
            int n = c / kq;
            int k = (c - n * kq) << 2;
            const float* src = (k < Kx) ? (Wih + (size_t)(n0 + n) * Kx + k)
                                        : (Whh + (size_t)(n0 + n) * Kh + (k - Kx));
            float4 v = *(const float4*)src;  // rows are 128B-multiples, k%4==0
            _Float16* d = Ws + (size_t)n * KT + k;
            d[0] = (_Float16)v.x; d[1] = (_Float16)v.y;
            d[2] = (_Float16)v.z; d[3] = (_Float16)v.w;
        }
    }
    __syncthreads();

    const int wv   = threadIdx.x >> 5;       // wave 0..7
    const int lane = threadIdx.x & 31;
    const int g    = lane >> 4;
    const int mn   = lane & 15;
    const int m0   = mBase + wv * 16;

    const float* arowX = Kx ? (X + (size_t)(m0 + mn) * Kx) : nullptr;
    const float* arowH = H + (size_t)(m0 + mn) * Kh;
    const _Float16* wsRow = Ws + (size_t)mn * KT;

    v8f acc0 = {}, acc1 = {}, acc2 = {}, acc3 = {};
    for (int k0 = 0; k0 < KT; k0 += 32) {
        // A fragment (direct from global; L2-resident activations)
        // Layout: lane l (m=l&15, g=l>>4): v[i<4] -> K=2i+8g ; v[i>=4] -> K=16+2(i-4)+8g
        const float* ar = (k0 < Kx) ? (arowX + k0) : (arowH + (k0 - Kx));
        __builtin_prefetch(ar + 32, 0, 1);   // gfx1250 global_prefetch path
        v16h a;
#pragma unroll
        for (int i = 0; i < 4; ++i) {
            int ka = 2 * i + 8 * g;
            a[2 * i]     = (_Float16)ar[ka];
            a[2 * i + 1] = (_Float16)ar[ka + 1];
            a[2 * i + 8] = (_Float16)ar[ka + 16];
            a[2 * i + 9] = (_Float16)ar[ka + 17];
        }
        // Load ALL four B fragments (one ds clause, one wait) ...
        v16h b0 = load_bfrag(wsRow + (size_t)(0 * 16) * KT + k0, g);
        v16h b1 = load_bfrag(wsRow + (size_t)(1 * 16) * KT + k0, g);
        v16h b2 = load_bfrag(wsRow + (size_t)(2 * 16) * KT + k0, g);
        v16h b3 = load_bfrag(wsRow + (size_t)(3 * 16) * KT + k0, g);
        // ... pin them live so the scheduler cannot re-serialize load->wmma pairs
        asm volatile("" : "+v"(a), "+v"(b0), "+v"(b1), "+v"(b2), "+v"(b3));
        // ... then issue the 4 WMMAs back-to-back (XDL burst)
        acc0 = __builtin_amdgcn_wmma_f32_16x16x32_f16(false, a, false, b0, (short)0, acc0, false, false);
        acc1 = __builtin_amdgcn_wmma_f32_16x16x32_f16(false, a, false, b1, (short)0, acc1, false, false);
        acc2 = __builtin_amdgcn_wmma_f32_16x16x32_f16(false, a, false, b2, (short)0, acc2, false, false);
        acc3 = __builtin_amdgcn_wmma_f32_16x16x32_f16(false, a, false, b3, (short)0, acc3, false, false);
    }

    // ---- store 16x64 strip (C/D: vgpr r, lane l: M=r+8g, N=l&15) ----
    v8f accs[4] = { acc0, acc1, acc2, acc3 };
#pragma unroll
    for (int s = 0; s < 4; ++s) {
        int nc = n0 + s * 16 + mn;
        float bv = bias ? bias[nc] : 0.f;
#pragma unroll
        for (int r = 0; r < 8; ++r)
            C[(size_t)(m0 + r + 8 * g) * N + nc] = accs[s][r] + bv;
    }
}

// ---------------- elementwise / small kernels ----------------
__global__ void k_embed(const float* __restrict__ traj2, const float* __restrict__ W_emb,
                        const float* __restrict__ b_emb, float* __restrict__ out) {
    int idx = blockIdx.x * blockDim.x + threadIdx.x;      // B*EIN exact
    int b = idx / EIN, e = idx - b * EIN;
    float v = traj2[b * 2] * W_emb[e * 2] + traj2[b * 2 + 1] * W_emb[e * 2 + 1] + b_emb[e];
    out[idx] = fmaxf(v, 0.f);
}

__global__ void k_lstm_update(const float* __restrict__ gates, const float* __restrict__ bih,
                              const float* __restrict__ bhh, float* __restrict__ h,
                              float* __restrict__ c, int Hd) {
    int idx = blockIdx.x * blockDim.x + threadIdx.x;      // B*Hd exact
    int b = idx / Hd, n = idx - b * Hd;
    const float* gr = gates + (size_t)b * 4 * Hd;
    float gi = gr[n]          + bih[n]          + bhh[n];
    float gf = gr[Hd + n]     + bih[Hd + n]     + bhh[Hd + n];
    float gg = gr[2 * Hd + n] + bih[2 * Hd + n] + bhh[2 * Hd + n];
    float go = gr[3 * Hd + n] + bih[3 * Hd + n] + bhh[3 * Hd + n];
    float cc = sigmoidf_(gf) * c[idx] + sigmoidf_(gi) * tanhf(gg);
    c[idx] = cc;
    h[idx] = sigmoidf_(go) * tanhf(cc);
}

__global__ void k_concat_noise(const float* __restrict__ h_enc, const float* __restrict__ noise,
                               float* __restrict__ h_dec) {
    int idx = blockIdx.x * blockDim.x + threadIdx.x;      // B*DHID exact
    int b = idx / DHID, n = idx - b * DHID;
    h_dec[idx] = (n < EHID) ? h_enc[(size_t)b * EHID + n]
                            : noise[(b / NPER) * NOISE + (n - EHID)];
}

__global__ void k_out(const float* __restrict__ h, const float* __restrict__ Wout,
                      const float* __restrict__ bout, float* __restrict__ traj,
                      float* __restrict__ outstep) {
    int idx = blockIdx.x * blockDim.x + threadIdx.x;      // B*2 exact
    int b = idx >> 1, d = idx & 1;
    const float* hr = h + (size_t)b * DHID;
    const float* wr = Wout + (size_t)d * DHID;
    float acc = bout[d];
    for (int k = 0; k < DHID; ++k) acc += hr[k] * wr[k];
    traj[idx]    = acc;
    outstep[idx] = acc;
}

// ---------------- signed-graph module ----------------
__global__ void k_scores(const float* __restrict__ embG, const float* __restrict__ Watt,
                         float* __restrict__ s_oth, float* __restrict__ s_self) {
    int b = blockIdx.x * blockDim.x + threadIdx.x;        // B exact
    const float* er = embG + (size_t)b * GEMB;
    float so = 0.f, ss = 0.f;
    for (int e = 0; e < GEMB; ++e) { so += er[e] * Watt[e]; ss += er[e] * Watt[GEMB + e]; }
    s_oth[b] = so; s_self[b] = ss;
}

__global__ void k_edge_softmax(const float* __restrict__ s_oth, const float* __restrict__ s_self,
                               const float* __restrict__ b_att, float* __restrict__ eraw) {
    __shared__ float red[256];
    int col = blockIdx.x;                  // GRP*NPER
    int g = col / NPER, j = col - g * NPER;
    int t = threadIdx.x;
    bool act = t < NPER;
    float v = -1e30f;
    if (act) v = leakyf_(s_oth[g * NPER + j] + s_self[g * NPER + t] + b_att[0]);
    float m  = blockReduceMax(v, red);
    float ex = act ? __expf(v - m) : 0.f;
    float sm = blockReduceSum(ex, red);
    if (act) eraw[(size_t)g * NPER * NPER + (size_t)t * NPER + j] = ex / sm;
}

__global__ void k_symmetrize(const float* __restrict__ eraw, float* __restrict__ edge) {
    int idx = blockIdx.x * blockDim.x + threadIdx.x;      // GRP*NPER*NPER exact
    int g = idx / (NPER * NPER);
    int r = idx - g * NPER * NPER;
    int i = r / NPER, j = r - i * NPER;
    size_t base = (size_t)g * NPER * NPER;
    edge[idx] = (i <= j) ? eraw[base + (size_t)i * NPER + j]
                         : eraw[base + (size_t)j * NPER + i];
}

__global__ void k_rowsum(const float* __restrict__ edge, float* __restrict__ dvec) {
    __shared__ float red[256];
    int row = blockIdx.x;                  // GRP*NPER
    int g = row / NPER, i = row - g * NPER;
    int t = threadIdx.x;
    float v = (t < NPER) ? edge[(size_t)g * NPER * NPER + (size_t)i * NPER + t] : 0.f;
    float s = blockReduceSum(v, red);
    if (t == 0) dvec[row] = s;
}

__global__ void k_lsym(const float* __restrict__ edge, const float* __restrict__ dvec,
                       float* __restrict__ lsym) {
    int idx = blockIdx.x * blockDim.x + threadIdx.x;
    int g = idx / (NPER * NPER);
    int r = idx - g * NPER * NPER;
    int i = r / NPER, j = r - i * NPER;
    float di = dvec[g * NPER + i], dj = dvec[g * NPER + j];
    float v = ((i == j) ? di : 0.f) - edge[idx];
    lsym[idx] = v * rsqrtf(fmaxf(di, 1e-12f)) * rsqrtf(fmaxf(dj, 1e-12f));
}

// Two smallest eigenvectors of L_sym via deflated power iteration on (2I - L_sym),
// then row-normalize (Hn). One workgroup per group; LDS-resident vectors.
__global__ void k_eigen2(const float* __restrict__ lsym, float* __restrict__ Hv) {
    __shared__ float v1s[NPER], v2s[NPER], red[256];
    const int g = blockIdx.x, t = threadIdx.x;
    const float* A = lsym + (size_t)g * NPER * NPER;
    const float inv0 = 0.07216878f;        // 1/sqrt(192)
    if (t < NPER) v1s[t] = inv0;
    __syncthreads();
    for (int it = 0; it < 64; ++it) {
        float acc = 0.f;
        if (t < NPER) {
            const float* row = A + (size_t)t * NPER;
            for (int j = 0; j < NPER; ++j) acc += row[j] * v1s[j];
            acc = 2.f * v1s[t] - acc;
        }
        float ss = blockReduceSum((t < NPER) ? acc * acc : 0.f, red);
        float inv = rsqrtf(fmaxf(ss, 1e-20f));
        if (t < NPER) v1s[t] = acc * inv;
        __syncthreads();
    }
    if (t < NPER) v2s[t] = (t & 1) ? -inv0 : inv0;
    __syncthreads();
    for (int it = 0; it < 64; ++it) {
        float acc = 0.f;
        if (t < NPER) {
            const float* row = A + (size_t)t * NPER;
            for (int j = 0; j < NPER; ++j) acc += row[j] * v2s[j];
            acc = 2.f * v2s[t] - acc;
        }
        float dt = blockReduceSum((t < NPER) ? acc * v1s[t] : 0.f, red);
        if (t < NPER) acc -= dt * v1s[t];
        float ss = blockReduceSum((t < NPER) ? acc * acc : 0.f, red);
        float inv = rsqrtf(fmaxf(ss, 1e-20f));
        if (t < NPER) v2s[t] = acc * inv;
        __syncthreads();
    }
    if (t < NPER) {
        float hx = v1s[t], hy = v2s[t];
        float rn = rsqrtf(fmaxf(hx * hx + hy * hy, 1e-20f));
        Hv[(size_t)(g * NPER + t) * 2 + 0] = hx * rn;
        Hv[(size_t)(g * NPER + t) * 2 + 1] = hy * rn;
    }
}

// k-means, k=96, 10 iters, 2-D points, LDS resident. One workgroup per group.
__global__ void k_kmeans(const float* __restrict__ Hv, int* __restrict__ labels) {
    __shared__ float px[NPER], py[NPER], cx[NPER / 2], cy[NPER / 2];
    __shared__ int lab[NPER];
    const int g = blockIdx.x, t = threadIdx.x;
    const int K = NPER / 2;                // 96
    if (t < NPER) { px[t] = Hv[(size_t)(g * NPER + t) * 2]; py[t] = Hv[(size_t)(g * NPER + t) * 2 + 1]; }
    __syncthreads();
    if (t < K) { cx[t] = px[t]; cy[t] = py[t]; }
    __syncthreads();
    for (int iter = 0; iter < 10; ++iter) {
        if (t < NPER) {
            float bx = px[t], by = py[t];
            int best = 0; float bd = 3.402823e38f;
            for (int k = 0; k < K; ++k) {
                float dx = bx - cx[k], dy = by - cy[k];
                float d2 = dx * dx + dy * dy;
                if (d2 < bd) { bd = d2; best = k; }
            }
            lab[t] = best;
        }
        __syncthreads();
        if (t < K) {
            float sx = 0.f, sy = 0.f; int cnt = 0;
            for (int j = 0; j < NPER; ++j)
                if (lab[j] == t) { sx += px[j]; sy += py[j]; ++cnt; }
            if (cnt > 0) { cx[t] = sx / (float)cnt; cy[t] = sy / (float)cnt; }
            else         { cx[t] = px[t]; cy[t] = py[t]; }
        }
        __syncthreads();
    }
    if (t < NPER) {
        float bx = px[t], by = py[t];
        int best = 0; float bd = 3.402823e38f;
        for (int k = 0; k < K; ++k) {
            float dx = bx - cx[k], dy = by - cy[k];
            float d2 = dx * dx + dy * dy;
            if (d2 < bd) { bd = d2; best = k; }
        }
        labels[g * NPER + t] = best;
    }
}

__global__ void k_alphas(const float* __restrict__ edge, const int* __restrict__ labels,
                         float* __restrict__ posa, float* __restrict__ nega) {
    __shared__ float red[256];
    int col = blockIdx.x;
    int g = col / NPER, j = col - g * NPER;
    int t = threadIdx.x;
    bool act = t < NPER;
    const float NEGC = -1e8f;
    float pv = -1e30f, nv = -1e30f;
    size_t eidx = (size_t)g * NPER * NPER + (size_t)t * NPER + j;
    int lj = labels[g * NPER + j];
    if (act) {
        float ev  = edge[eidx];
        bool same = labels[g * NPER + t] == lj;
        pv = leakyf_(same ? ev : ev * NEGC);
        nv = leakyf_(same ? ev * NEGC : ev);
    }
    float pm = blockReduceMax(pv, red);
    float pe = act ? __expf(pv - pm) : 0.f;
    float ps = blockReduceSum(pe, red);
    float nm = blockReduceMax(nv, red);
    float ne = act ? __expf(nv - nm) : 0.f;
    float ns = blockReduceSum(ne, red);
    if (act) { posa[eidx] = pe / ps; nega[eidx] = ne / ns; }
}

__global__ void k_feat(const float* __restrict__ posa, const float* __restrict__ nega,
                       const float* __restrict__ node, float* __restrict__ hnew) {
    __shared__ float pa[NPER], na[NPER];
    int row = blockIdx.x;                  // BATCH
    int g = row / NPER, i = row - g * NPER;
    int t = threadIdx.x;                   // DHID threads
    size_t base = (size_t)g * NPER * NPER + (size_t)i * NPER;
    if (t < NPER) { pa[t] = posa[base + t]; na[t] = nega[base + t]; }
    __syncthreads();
    const float* nb = node + (size_t)g * NPER * DHID;
    float acc = 0.f;
    for (int j = 0; j < NPER; ++j) {
        float x = nb[(size_t)j * DHID + t];
        acc += fmaxf(pa[j] * x, 0.f) + fmaxf(na[j] * x, 0.f);
    }
    hnew[(size_t)row * DHID + t] = acc;
}

// ---------------- host driver ----------------
extern "C" void kernel_launch(void* const* d_in, const int* in_sizes, int n_in,
                              void* d_out, int out_size, void* d_ws, size_t ws_size,
                              hipStream_t stream) {
    const float* input_traj = (const float*)d_in[0];
    const float* enc_h0     = (const float*)d_in[1];
    const float* enc_c0     = (const float*)d_in[2];
    const float* noise_z    = (const float*)d_in[3];
    const float* W_emb      = (const float*)d_in[4];
    const float* b_emb      = (const float*)d_in[5];
    const float* Wih_e      = (const float*)d_in[6];
    const float* Whh_e      = (const float*)d_in[7];
    const float* bih_e      = (const float*)d_in[8];
    const float* bhh_e      = (const float*)d_in[9];
    const float* Wih_d      = (const float*)d_in[10];
    const float* Whh_d      = (const float*)d_in[11];
    const float* bih_d      = (const float*)d_in[12];
    const float* bhh_d      = (const float*)d_in[13];
    const float* W_out      = (const float*)d_in[14];
    const float* b_out      = (const float*)d_in[15];
    const float* W_ne       = (const float*)d_in[16];
    const float* b_ne       = (const float*)d_in[17];
    const float* W_att      = (const float*)d_in[18];
    const float* b_att      = (const float*)d_in[19];
    // d_in[20] (seq_start_end) implied by GRP/NPER constants

    float* dout = (float*)d_out;

    // ---- workspace carve (~24 MB of floats + labels) ----
    float* p = (float*)d_ws;
    float* h_enc = p; p += (size_t)BATCH * EHID;
    float* c_enc = p; p += (size_t)BATCH * EHID;
    float* h_dec = p; p += (size_t)BATCH * DHID;
    float* c_dec = p; p += (size_t)BATCH * DHID;
    float* hnew  = p; p += (size_t)BATCH * DHID;
    float* gates = p; p += (size_t)BATCH * 4 * DHID;
    float* xemb  = p; p += (size_t)BATCH * EIN;
    float* embG  = p; p += (size_t)BATCH * GEMB;
    float* traj  = p; p += (size_t)BATCH * 2;
    float* s_oth = p; p += BATCH;
    float* s_slf = p; p += BATCH;
    float* dvec  = p; p += BATCH;
    float* Hvec  = p; p += (size_t)BATCH * 2;
    float* eraw  = p; p += (size_t)GRP * NPER * NPER;
    float* edge  = p; p += (size_t)GRP * NPER * NPER;
    float* lsym  = p; p += (size_t)GRP * NPER * NPER;
    float* posa  = p; p += (size_t)GRP * NPER * NPER;
    float* nega  = p; p += (size_t)GRP * NPER * NPER;
    int*   labels = (int*)p;

    const size_t fB = sizeof(float);

    // ---- init encoder state ----
    hipMemcpyAsync(h_enc, enc_h0, (size_t)BATCH * EHID * fB, hipMemcpyDeviceToDevice, stream);
    hipMemcpyAsync(c_enc, enc_c0, (size_t)BATCH * EHID * fB, hipMemcpyDeviceToDevice, stream);

    // ---- encoder: 8 LSTM steps (blocked WMMA gate GEMMs) ----
    const int gatesN_e = 4 * EHID;   // 1024
    const int gatesN_d = 4 * DHID;   // 1280
    for (int t = 0; t < OBS; ++t) {
        k_embed<<<BATCH * EIN / 256, 256, 0, stream>>>(
            input_traj + (size_t)t * BATCH * 2, W_emb, b_emb, xemb);
        dim3 gd(gatesN_e / 64, BATCH / 128);
        k_gemm_block<<<gd, 256, 0, stream>>>(
            xemb, EIN, Wih_e, h_enc, EHID, Whh_e, nullptr, gates, BATCH, gatesN_e);
        k_lstm_update<<<BATCH * EHID / 256, 256, 0, stream>>>(
            gates, bih_e, bhh_e, h_enc, c_enc, EHID);
    }

    // ---- decoder init: h = [h_enc | z], c = 0, traj = input_traj[OBS-1] ----
    k_concat_noise<<<BATCH * DHID / 256, 256, 0, stream>>>(h_enc, noise_z, h_dec);
    hipMemsetAsync(c_dec, 0, (size_t)BATCH * DHID * fB, stream);
    hipMemcpyAsync(traj, input_traj + (size_t)(OBS - 1) * BATCH * 2, (size_t)BATCH * 2 * fB,
                   hipMemcpyDeviceToDevice, stream);

    // ---- decoder: 12 steps, signed-graph at i==4 ----
    for (int i = 0; i < PRED; ++i) {
        k_embed<<<BATCH * EIN / 256, 256, 0, stream>>>(traj, W_emb, b_emb, xemb);

        if (i == 4) {
            dim3 ge(GEMB / 64, BATCH / 128);
            k_gemm_block<<<ge, 256, 0, stream>>>(
                nullptr, 0, nullptr, h_dec, DHID, W_ne, b_ne, embG, BATCH, GEMB);
            k_scores<<<BATCH / 256, 256, 0, stream>>>(embG, W_att, s_oth, s_slf);
            k_edge_softmax<<<GRP * NPER, 256, 0, stream>>>(s_oth, s_slf, b_att, eraw);
            k_symmetrize<<<GRP * NPER * NPER / 256, 256, 0, stream>>>(eraw, edge);
            k_rowsum<<<GRP * NPER, 256, 0, stream>>>(edge, dvec);
            k_lsym<<<GRP * NPER * NPER / 256, 256, 0, stream>>>(edge, dvec, lsym);
            k_eigen2<<<GRP, 256, 0, stream>>>(lsym, Hvec);
            k_kmeans<<<GRP, 256, 0, stream>>>(Hvec, labels);
            k_alphas<<<GRP * NPER, 256, 0, stream>>>(edge, labels, posa, nega);
            k_feat<<<BATCH, DHID, 0, stream>>>(posa, nega, h_dec, hnew);
            hipMemcpyAsync(h_dec, hnew, (size_t)BATCH * DHID * fB,
                           hipMemcpyDeviceToDevice, stream);
        }

        dim3 gd(gatesN_d / 64, BATCH / 128);
        k_gemm_block<<<gd, 256, 0, stream>>>(
            xemb, EIN, Wih_d, h_dec, DHID, Whh_d, nullptr, gates, BATCH, gatesN_d);
        k_lstm_update<<<BATCH * DHID / 256, 256, 0, stream>>>(
            gates, bih_d, bhh_d, h_dec, c_dec, DHID);
        k_out<<<BATCH * 2 / 256, 256, 0, stream>>>(
            h_dec, W_out, b_out, traj, dout + (size_t)i * BATCH * 2);
    }
}
